// LEGNN_53790170415696
// MI455X (gfx1250) — matrix operand order repaired and checked by reading, not compiled, not run on hardware
//
#include <hip/hip_runtime.h>
#include <hip/hip_bf16.h>

// ---------------------------------------------------------------------------
// LEGNN forward for MI455X (gfx1250, wave32, WMMA).
// Dense GEMMs: v_wmma_f32_16x16x32_bf16 with bf16 hi/lo split (~f32 accuracy).
// Weights are stored transposed (B^T[Ncol][K]) so every lane's B fragment is
// one contiguous 32B load (2x global_load_b128) instead of 16 strided b16
// loads + v_mov_b16 packing. A fragments are two contiguous 16B vector loads.
// Sparse aggregation exploits adj sparsity (CSR built once per launch).
// ---------------------------------------------------------------------------

#define NN      8000
#define NFEAT   512
#define NHID    128
#define NCLASS  64
#define NTRAIN  500
#define CSR_CAP 64
#define ALPHA   0.1f
#define EPSV    1e-12f

typedef __bf16 bf16_t;
typedef __attribute__((ext_vector_type(16))) __bf16 v16bf;
typedef __attribute__((ext_vector_type(8)))  __bf16 v8bf;
typedef __attribute__((ext_vector_type(8)))  float  v8f;

// ---------------------------------------------------------------------------
// f32 -> bf16 hi/lo split (hi + lo reproduces f32 to ~2^-30 relative)
// ---------------------------------------------------------------------------
__global__ void k_split_bf16(const float* __restrict__ src,
                             bf16_t* __restrict__ hi, bf16_t* __restrict__ lo,
                             int n) {
    int i = blockIdx.x * blockDim.x + threadIdx.x;
    if (i < n) {
        float x  = src[i];
        bf16_t h = (bf16_t)x;
        hi[i] = h;
        lo[i] = (bf16_t)(x - (float)h);
    }
}

// Same split, but writes the transposed layout [Ncol][K] (weights only; tiny).
__global__ void k_split_bf16_T(const float* __restrict__ src,   // [K][Ncol]
                               bf16_t* __restrict__ hiT,        // [Ncol][K]
                               bf16_t* __restrict__ loT,
                               int K, int Ncol) {
    int i = blockIdx.x * blockDim.x + threadIdx.x;
    if (i < K * Ncol) {
        int r = i / Ncol, c = i % Ncol;
        float x  = src[i];
        bf16_t h = (bf16_t)x;
        hiT[(size_t)c * K + r] = h;
        loT[(size_t)c * K + r] = (bf16_t)(x - (float)h);
    }
}

// ---------------------------------------------------------------------------
// C[M,Ncol] = act(A[M,K] @ B[K,Ncol] + bias); B passed transposed [Ncol][K].
// One wave per 16x16 C tile.
//   A frag (16-bit A 16x32): lane L, half=L/16, row=m0+L%16,
//     elements 0..7  = K k0+16*half .. +7       (contiguous)
//     elements 8..15 = K k0+16*half+8 .. +7     (contiguous)
//   B frag (16-bit B 32x16): lane L, col=n0+L%16,
//     elements 0..15 = K k0+16*half .. +15      (contiguous in B^T)
//   C frag: VGPR r -> row m0 + r + 8*half, col n0 + L%16.
// ---------------------------------------------------------------------------
template <bool RELU>
__global__ __launch_bounds__(32)
void k_wmma_gemm(const bf16_t* __restrict__ Ahi, const bf16_t* __restrict__ Alo,
                 const bf16_t* __restrict__ BThi, const bf16_t* __restrict__ BTlo,
                 const float* __restrict__ bias, float* __restrict__ C,
                 int M, int K, int Ncol) {
    const int lane = threadIdx.x & 31;
    const int half = lane >> 4;   // 0 | 1
    const int l16  = lane & 15;
    const int m0   = blockIdx.x << 4;
    const int n0   = blockIdx.y << 4;

    const bf16_t* aPh = Ahi  + (size_t)(m0 + l16) * K;
    const bf16_t* aPl = Alo  + (size_t)(m0 + l16) * K;
    const bf16_t* bPh = BThi + (size_t)(n0 + l16) * K;
    const bf16_t* bPl = BTlo + (size_t)(n0 + l16) * K;

    v8f acc = {};
    for (int k0 = 0; k0 < K; k0 += 32) {
        const v8bf a0h = *(const v8bf*)(aPh + k0 + 16 * half);
        const v8bf a1h = *(const v8bf*)(aPh + k0 + 16 * half + 8);
        const v8bf a0l = *(const v8bf*)(aPl + k0 + 16 * half);
        const v8bf a1l = *(const v8bf*)(aPl + k0 + 16 * half + 8);
        const v16bf ah = __builtin_shufflevector(a0h, a1h,
            0, 1, 2, 3, 4, 5, 6, 7, 8, 9, 10, 11, 12, 13, 14, 15);
        const v16bf al = __builtin_shufflevector(a0l, a1l,
            0, 1, 2, 3, 4, 5, 6, 7, 8, 9, 10, 11, 12, 13, 14, 15);
        const v16bf bh = *(const v16bf*)(bPh + k0 + 16 * half);
        const v16bf bl = *(const v16bf*)(bPl + k0 + 16 * half);

        // hi*hi + hi*lo + lo*hi  (lo*lo term ~2^-32, dropped)
        acc = __builtin_amdgcn_wmma_f32_16x16x32_bf16(false, ah, false, bh,
                                                      (short)0, acc, false, false);
        acc = __builtin_amdgcn_wmma_f32_16x16x32_bf16(false, ah, false, bl,
                                                      (short)0, acc, false, false);
        acc = __builtin_amdgcn_wmma_f32_16x16x32_bf16(false, al, false, bh,
                                                      (short)0, acc, false, false);
    }

    const float bv = bias[n0 + l16];
#pragma unroll
    for (int r = 0; r < 8; ++r) {
        const int row = m0 + r + 8 * half;
        float v = acc[r] + bv;
        if (RELU) v = fmaxf(v, 0.0f);
        C[(size_t)row * Ncol + n0 + l16] = v;
    }
}

// ---------------------------------------------------------------------------
// Pseudo[idx[t]] += 0.1 * y_label[idx[t]]  — single block, sequential over t
// (deterministic even with duplicated indices).
// ---------------------------------------------------------------------------
__global__ __launch_bounds__(NCLASS)
void k_inject(float* __restrict__ Pseudo, const int* __restrict__ idx,
              const float* __restrict__ y_label, int ntrain) {
    const int c = threadIdx.x;
    for (int t = 0; t < ntrain; ++t) {
        const int r = idx[t];
        Pseudo[r * NCLASS + c] += 0.1f * y_label[r * NCLASS + c];
    }
}

// ---------------------------------------------------------------------------
// Row softmax / log-softmax over NCLASS=64, one 64-thread block per row.
// ---------------------------------------------------------------------------
__global__ __launch_bounds__(NCLASS)
void k_softmax64(const float* __restrict__ in, float* __restrict__ out) {
    __shared__ float red[NCLASS];
    const int r = blockIdx.x, c = threadIdx.x;
    const float x = in[r * NCLASS + c];
    red[c] = x; __syncthreads();
    for (int s = 32; s > 0; s >>= 1) { if (c < s) red[c] = fmaxf(red[c], red[c + s]); __syncthreads(); }
    const float m = red[0]; __syncthreads();
    const float e = expf(x - m);
    red[c] = e; __syncthreads();
    for (int s = 32; s > 0; s >>= 1) { if (c < s) red[c] += red[c + s]; __syncthreads(); }
    out[r * NCLASS + c] = e / red[0];
}

__global__ __launch_bounds__(NCLASS)
void k_logsoftmax64(const float* __restrict__ in, float* __restrict__ out) {
    __shared__ float red[NCLASS];
    const int r = blockIdx.x, c = threadIdx.x;
    const float x = in[r * NCLASS + c];
    red[c] = x; __syncthreads();
    for (int s = 32; s > 0; s >>= 1) { if (c < s) red[c] = fmaxf(red[c], red[c + s]); __syncthreads(); }
    const float m = red[0]; __syncthreads();
    const float e = expf(x - m);
    red[c] = e; __syncthreads();
    for (int s = 32; s > 0; s >>= 1) { if (c < s) red[c] += red[c + s]; __syncthreads(); }
    out[r * NCLASS + c] = (x - m) - logf(red[0]);
}

// ---------------------------------------------------------------------------
// Build per-row nonzero column lists from dense adj. One wave per row;
// ballot compaction keeps column order ascending (deterministic accumulation
// order downstream). 256 MB streamed once -> ~11 us at 23.3 TB/s.
// ---------------------------------------------------------------------------
__global__ __launch_bounds__(32)
void k_build_csr(const float* __restrict__ adj, int* __restrict__ colidx,
                 int* __restrict__ rowcnt) {
    const int row  = blockIdx.x;
    const int lane = threadIdx.x;
    const float* arow = adj + (size_t)row * NN;
    int cnt = 0;
    for (int c0 = 0; c0 < NN; c0 += 32) {
        if (c0 + 2048 < NN) __builtin_prefetch(arow + c0 + 2048, 0, 0);
        const float v = arow[c0 + lane];
        const unsigned m = (unsigned)__ballot(v != 0.0f);
        if (v != 0.0f) {
            const int pos = cnt + __popc(m & ((1u << lane) - 1u));
            if (pos < CSR_CAP) colidx[row * CSR_CAP + pos] = c0 + lane;
        }
        cnt += __popc(m);
    }
    if (lane == 0) rowcnt[row] = cnt < CSR_CAP ? cnt : CSR_CAP;
}

// ---------------------------------------------------------------------------
// One propagation layer, fused:
//   w_ij = y_hat[i] . y_hat[j]         (adj values are exactly 1.0)
//   support = (sum_j w_ij * li[j]) / max(sum_j w_ij, eps)   [w >= 0 -> L1=sum]
//   li' = 0.9*support + 0.1*h0 ; li' /= max(||li'||_2, eps)
// Block = 128 threads (one per hidden channel). The 64-wide dot is computed
// redundantly per thread: deterministic, no intra-loop syncs.
// ---------------------------------------------------------------------------
__global__ __launch_bounds__(NHID)
void k_aggregate(const float* __restrict__ yhat, const float* __restrict__ li_in,
                 const float* __restrict__ h0, const int* __restrict__ colidx,
                 const int* __restrict__ rowcnt, float* __restrict__ li_out) {
    __shared__ float yi[NCLASS];
    __shared__ float red[NHID];
    const int i = blockIdx.x, t = threadIdx.x;
    if (t < NCLASS) yi[t] = yhat[i * NCLASS + t];
    __syncthreads();

    const int  cnt  = rowcnt[i];
    const int* cols = colidx + i * CSR_CAP;
    float acc = 0.0f, rowsum = 0.0f;
    for (int e = 0; e < cnt; ++e) {
        const int j = cols[e];
        const float* yj = yhat + j * NCLASS;
        float w = 0.0f;
#pragma unroll 16
        for (int c = 0; c < NCLASS; ++c) w += yi[c] * yj[c];
        rowsum += w;
        acc += w * li_in[j * NHID + t];
    }
    float li = 0.9f * (acc / fmaxf(rowsum, EPSV)) + ALPHA * h0[i * NHID + t];

    red[t] = li * li; __syncthreads();
    for (int s = 64; s > 0; s >>= 1) { if (t < s) red[t] += red[t + s]; __syncthreads(); }
    li_out[i * NHID + t] = li / fmaxf(sqrtf(red[0]), EPSV);
}

// ---------------------------------------------------------------------------
// Launcher
// ---------------------------------------------------------------------------
extern "C" void kernel_launch(void* const* d_in, const int* in_sizes, int n_in,
                              void* d_out, int out_size, void* d_ws, size_t ws_size,
                              hipStream_t stream) {
    const float* x        = (const float*)d_in[0];
    const float* adj      = (const float*)d_in[1];
    const float* y_label  = (const float*)d_in[2];
    const int*   idx_tr   = (const int*)  d_in[3];
    const float* W0       = (const float*)d_in[4];
    const float* b0       = (const float*)d_in[5];
    const float* W1       = (const float*)d_in[6];
    const float* b1       = (const float*)d_in[7];
    const float* W2       = (const float*)d_in[8];
    const float* b2       = (const float*)d_in[9];

    float* out_ls  = (float*)d_out;                       // [N, NCLASS] log_softmax
    float* Pseudo  = (float*)d_out + (size_t)NN * NCLASS; // [N, NCLASS]

    // ---- workspace carve (32B aligned segments: v16bf loads need 32B) ----
    char* w = (char*)d_ws;
    auto carve = [&](size_t bytes) { void* p = w; w += (bytes + 31) & ~size_t(31); return p; };
    float*  h      = (float*) carve((size_t)NN * NHID   * 4);
    float*  yhat   = (float*) carve((size_t)NN * NCLASS * 4);
    float*  liA    = (float*) carve((size_t)NN * NHID   * 4);
    float*  liB    = (float*) carve((size_t)NN * NHID   * 4);
    float*  outraw = (float*) carve((size_t)NN * NCLASS * 4);
    bf16_t* Xhi    = (bf16_t*)carve((size_t)NN * NFEAT  * 2);
    bf16_t* Xlo    = (bf16_t*)carve((size_t)NN * NFEAT  * 2);
    bf16_t* W0hi   = (bf16_t*)carve((size_t)NFEAT * NHID * 2);   // transposed [NHID][NFEAT]
    bf16_t* W0lo   = (bf16_t*)carve((size_t)NFEAT * NHID * 2);
    bf16_t* Hhi    = (bf16_t*)carve((size_t)NN * NHID   * 2);
    bf16_t* Hlo    = (bf16_t*)carve((size_t)NN * NHID   * 2);
    bf16_t* W1hi   = (bf16_t*)carve((size_t)NHID * NCLASS * 2);  // transposed [NCLASS][NHID]
    bf16_t* W1lo   = (bf16_t*)carve((size_t)NHID * NCLASS * 2);
    bf16_t* Lhi    = (bf16_t*)carve((size_t)NN * NHID   * 2);
    bf16_t* Llo    = (bf16_t*)carve((size_t)NN * NHID   * 2);
    bf16_t* W2hi   = (bf16_t*)carve((size_t)NHID * NCLASS * 2);  // transposed [NCLASS][NHID]
    bf16_t* W2lo   = (bf16_t*)carve((size_t)NHID * NCLASS * 2);
    int*    colidx = (int*)   carve((size_t)NN * CSR_CAP * 4);
    int*    rowcnt = (int*)   carve((size_t)NN * 4);

    const int T = 256;
    auto blocks = [](int n, int t) { return (n + t - 1) / t; };

    // 1) split f32 inputs to bf16 hi/lo (weights transposed for wide B loads)
    k_split_bf16<<<blocks(NN * NFEAT, T),      T, 0, stream>>>(x,  Xhi,  Xlo,  NN * NFEAT);
    k_split_bf16_T<<<blocks(NFEAT * NHID, T),  T, 0, stream>>>(W0, W0hi, W0lo, NFEAT, NHID);
    k_split_bf16_T<<<blocks(NHID * NCLASS, T), T, 0, stream>>>(W1, W1hi, W1lo, NHID, NCLASS);
    k_split_bf16_T<<<blocks(NHID * NCLASS, T), T, 0, stream>>>(W2, W2hi, W2lo, NHID, NCLASS);

    // 2) h = relu(x @ W0 + b0)
    k_wmma_gemm<true><<<dim3(NN / 16, NHID / 16), 32, 0, stream>>>(
        Xhi, Xlo, W0hi, W0lo, b0, h, NN, NFEAT, NHID);

    // 3) Pseudo = h @ W1 + b1  (written directly into d_out second half)
    k_split_bf16<<<blocks(NN * NHID, T), T, 0, stream>>>(h, Hhi, Hlo, NN * NHID);
    k_wmma_gemm<false><<<dim3(NN / 16, NCLASS / 16), 32, 0, stream>>>(
        Hhi, Hlo, W1hi, W1lo, b1, Pseudo, NN, NHID, NCLASS);

    // 4) label injection (deterministic sequential), then y_hat = softmax(Pseudo)
    k_inject<<<1, NCLASS, 0, stream>>>(Pseudo, idx_tr, y_label, NTRAIN);
    k_softmax64<<<NN, NCLASS, 0, stream>>>(Pseudo, yhat);

    // 5) sparse structure of adj (single 256 MB streaming pass)
    k_build_csr<<<NN, 32, 0, stream>>>(adj, colidx, rowcnt);

    // 6) two propagation layers (li0 = h, h0 = h)
    k_aggregate<<<NN, NHID, 0, stream>>>(yhat, h,   h, colidx, rowcnt, liA);
    k_aggregate<<<NN, NHID, 0, stream>>>(yhat, liA, h, colidx, rowcnt, liB);

    // 7) out = li @ W2 + b2 ; log_softmax rows -> d_out first half
    k_split_bf16<<<blocks(NN * NHID, T), T, 0, stream>>>(liB, Lhi, Llo, NN * NHID);
    k_wmma_gemm<false><<<dim3(NN / 16, NCLASS / 16), 32, 0, stream>>>(
        Lhi, Llo, W2hi, W2lo, b2, outraw, NN, NHID, NCLASS);
    k_logsoftmax64<<<NN, NCLASS, 0, stream>>>(outraw, out_ls);

    (void)in_sizes; (void)n_in; (void)out_size; (void)ws_size;
}
